// GCNClassifier_3358664425871
// MI455X (gfx1250) — matrix operand order repaired and checked by reading, not compiled, not run on hardware
//
#include <hip/hip_runtime.h>
#include <stdint.h>

typedef __bf16 bf16t;
typedef __attribute__((ext_vector_type(16))) __bf16 v16bf;
typedef __attribute__((ext_vector_type(8)))  __bf16 bf16x8;
typedef __attribute__((ext_vector_type(4)))  __bf16 bf16x4;
typedef __attribute__((ext_vector_type(8)))  float  v8f;

// ---------------------------------------------------------------- utilities

__global__ void k_fill(float* __restrict__ p, float v, int n) {
  int i = blockIdx.x * blockDim.x + threadIdx.x;
  if (i < n) p[i] = v;
}

// float -> bf16, 4 elements per thread
__global__ void k_f2bf(const float* __restrict__ in, bf16t* __restrict__ out, int n4) {
  int i = blockIdx.x * blockDim.x + threadIdx.x;
  if (i < n4) {
    float4 v = ((const float4*)in)[i];
    bf16x4 o;
    o[0] = (bf16t)v.x; o[1] = (bf16t)v.y; o[2] = (bf16t)v.z; o[3] = (bf16t)v.w;
    ((bf16x4*)out)[i] = o;
  }
}

static __device__ __forceinline__ void atomicMaxF(float* addr, float val) {
  unsigned int* ua = (unsigned int*)addr;
  unsigned int old = *(volatile unsigned int*)ua;
  while (__uint_as_float(old) < val) {
    unsigned int assumed = old;
    old = atomicCAS(ua, assumed, __float_as_uint(val));
    if (old == assumed) break;
  }
}

// ------------------------------------------------------- weight fragment pack
// W: [128][ncols] row-major f32 -> Bpack[kt][nt][v][lane] dwords, each dword =
// {bf16(W[k+1][n]) , bf16(W[k][n])} with k = kt*32 + 16*(lane>>4) + 2v,
// n = nt*16 + (lane&15).  Contiguous k order (k = 16*half + s), same as the A loads in k_gemm.
__global__ void k_pack_w(const float* __restrict__ W, uint32_t* __restrict__ Bp, int ncols) {
  int idx = blockIdx.x * blockDim.x + threadIdx.x;
  int ntiles = ncols >> 4;
  int total = 4 * ntiles * 8 * 32;
  if (idx >= total) return;
  int lane = idx & 31;
  int v    = (idx >> 5) & 7;
  int nt   = (idx >> 8) % ntiles;
  int kt   = (idx >> 8) / ntiles;
  int half = lane >> 4;
  int n = nt * 16 + (lane & 15);
  int k = kt * 32 + 16 * half + 2 * v;
  bf16t b0 = (bf16t)W[(size_t)k * ncols + n];
  bf16t b1 = (bf16t)W[(size_t)(k + 1) * ncols + n];
  uint32_t u0 = (uint32_t)*(const unsigned short*)&b0;
  uint32_t u1 = (uint32_t)*(const unsigned short*)&b1;
  Bp[idx] = u0 | (u1 << 16);
}

// ----------------------------------------------------------------- WMMA GEMM
// C[M, NT*16] = A[M,128](bf16) x W(packed), optional bias/relu fused on store.
// One wave per 16-row stripe; K=128 as 4 steps of 32; NT column tiles.
template <int NT>
__global__ void __launch_bounds__(256) k_gemm(const bf16t* __restrict__ A,
                                              const uint32_t* __restrict__ Bp,
                                              const float* __restrict__ bias,
                                              float* __restrict__ C,
                                              int M, int relu) {
  const int wave = blockIdx.x * 8 + (threadIdx.x >> 5);
  const int lane = threadIdx.x & 31;
  const int row0 = wave << 4;
  if (row0 >= M) return;
  const int m = lane & 15;
  const int half = lane >> 4;
  const bf16t* arow = A + (size_t)(row0 + m) * 128;

  const v8f vzero = {0.f, 0.f, 0.f, 0.f, 0.f, 0.f, 0.f, 0.f};
  v8f acc[NT];
#pragma unroll
  for (int t = 0; t < NT; ++t) acc[t] = vzero;

#pragma unroll
  for (int kt = 0; kt < 4; ++kt) {
    // A fragment: contiguous k = 16*half + s, as in the B pack -> two contiguous 16B chunks per lane
    union { v16bf v; bf16x8 h[2]; } au;
    au.h[0] = *(const bf16x8*)(arow + kt * 32 + 16 * half);
    au.h[1] = *(const bf16x8*)(arow + kt * 32 + 16 * half + 8);
#pragma unroll
    for (int nt = 0; nt < NT; ++nt) {
      union { v16bf v; uint32_t u[8]; } bu;
      const uint32_t* bp = Bp + (size_t)((kt * NT + nt) * 8) * 32 + lane;
#pragma unroll
      for (int r = 0; r < 8; ++r) bu.u[r] = bp[r * 32];
      acc[nt] = __builtin_amdgcn_wmma_f32_16x16x32_bf16(
          false, au.v, false, bu.v, (short)0, acc[nt], false, false);
    }
  }

  const int ldc = NT * 16;
#pragma unroll
  for (int nt = 0; nt < NT; ++nt) {
    int n = nt * 16 + m;
    float bv = bias ? bias[n] : 0.0f;
#pragma unroll
    for (int r = 0; r < 8; ++r) {
      int row = row0 + r + 8 * half;   // C/D f32 layout: VGPR r <-> M = r + 8*half
      float val = acc[nt][r] + bv;
      if (relu) val = fmaxf(val, 0.0f);
      C[(size_t)row * ldc + n] = val;
    }
  }
}

// ----------------------------------------------------------------- graph prep
__global__ void k_deg(const int* __restrict__ dst, float* __restrict__ deg, int E) {
  int e = blockIdx.x * blockDim.x + threadIdx.x;
  if (e < E) atomicAdd(&deg[dst[e]], 1.0f);
}

__global__ void k_norm(const float* __restrict__ deg, float* __restrict__ dinv,
                       float* __restrict__ sfn, int n) {
  int i = blockIdx.x * blockDim.x + threadIdx.x;
  if (i < n) {
    float d = deg[i] + 1.0f;
    dinv[i] = rsqrtf(d);
    sfn[i]  = 1.0f / d;
  }
}

// One wave per edge: coalesced 512B gather from xw[src], 4 atomics per lane.
__global__ void __launch_bounds__(256) k_scatter(const int* __restrict__ src,
                                                 const int* __restrict__ dst,
                                                 const float* __restrict__ dinv,
                                                 const float* __restrict__ xw,
                                                 float* __restrict__ agg, int E) {
  int e = blockIdx.x * 8 + (threadIdx.x >> 5);
  if (e >= E) return;
  int lane = threadIdx.x & 31;
  int s = src[e], d = dst[e];
  float norm = dinv[s] * dinv[d];
  float4 v = *(const float4*)(xw + (size_t)s * 128 + lane * 4);
  float* ap = agg + (size_t)d * 128 + lane * 4;
  atomicAdd(ap + 0, v.x * norm);
  atomicAdd(ap + 1, v.y * norm);
  atomicAdd(ap + 2, v.z * norm);
  atomicAdd(ap + 3, v.w * norm);
}

// h = relu(LN(agg + xw*selfn + b)) + h   (one wave per node, shfl reductions)
__global__ void __launch_bounds__(256) k_conv_post(
    const float* __restrict__ agg, const float* __restrict__ xw,
    const float* __restrict__ sfn, const float* __restrict__ bias,
    const float* __restrict__ gamma, const float* __restrict__ beta,
    float* __restrict__ h, int n) {
  int node = blockIdx.x * 8 + (threadIdx.x >> 5);
  if (node >= n) return;
  int lane = threadIdx.x & 31;
  size_t base = (size_t)node * 128 + lane * 4;
  float sn = sfn[node];
  float4 a = *(const float4*)(agg + base);
  float4 w = *(const float4*)(xw + base);
  float4 b = *(const float4*)(bias + lane * 4);
  float v[4] = {a.x + w.x * sn + b.x, a.y + w.y * sn + b.y,
                a.z + w.z * sn + b.z, a.w + w.w * sn + b.w};
  float s = v[0] + v[1] + v[2] + v[3];
#pragma unroll
  for (int off = 16; off >= 1; off >>= 1) s += __shfl_xor(s, off, 32);
  float mean = s * (1.0f / 128.0f);
  float var = 0.f;
#pragma unroll
  for (int i = 0; i < 4; ++i) { float d = v[i] - mean; var += d * d; }
#pragma unroll
  for (int off = 16; off >= 1; off >>= 1) var += __shfl_xor(var, off, 32);
  float inv = rsqrtf(var * (1.0f / 128.0f) + 1e-5f);
  float4 g4 = *(const float4*)(gamma + lane * 4);
  float4 bt = *(const float4*)(beta + lane * 4);
  float4 hh = *(const float4*)(h + base);
  float4 o;
  o.x = fmaxf((v[0] - mean) * inv * g4.x + bt.x, 0.f) + hh.x;
  o.y = fmaxf((v[1] - mean) * inv * g4.y + bt.y, 0.f) + hh.y;
  o.z = fmaxf((v[2] - mean) * inv * g4.z + bt.z, 0.f) + hh.z;
  o.w = fmaxf((v[3] - mean) * inv * g4.w + bt.w, 0.f) + hh.w;
  *(float4*)(h + base) = o;
}

// out = relu(LN(in + b))  (node head LN, one wave per node)
__global__ void __launch_bounds__(256) k_ln_relu(
    const float* __restrict__ in, const float* __restrict__ bias,
    const float* __restrict__ gamma, const float* __restrict__ beta,
    float* __restrict__ out, int n) {
  int node = blockIdx.x * 8 + (threadIdx.x >> 5);
  if (node >= n) return;
  int lane = threadIdx.x & 31;
  size_t base = (size_t)node * 128 + lane * 4;
  float4 a = *(const float4*)(in + base);
  float4 b = *(const float4*)(bias + lane * 4);
  float v[4] = {a.x + b.x, a.y + b.y, a.z + b.z, a.w + b.w};
  float s = v[0] + v[1] + v[2] + v[3];
#pragma unroll
  for (int off = 16; off >= 1; off >>= 1) s += __shfl_xor(s, off, 32);
  float mean = s * (1.0f / 128.0f);
  float var = 0.f;
#pragma unroll
  for (int i = 0; i < 4; ++i) { float d = v[i] - mean; var += d * d; }
#pragma unroll
  for (int off = 16; off >= 1; off >>= 1) var += __shfl_xor(var, off, 32);
  float inv = rsqrtf(var * (1.0f / 128.0f) + 1e-5f);
  float4 g4 = *(const float4*)(gamma + lane * 4);
  float4 bt = *(const float4*)(beta + lane * 4);
  float4 o;
  o.x = fmaxf((v[0] - mean) * inv * g4.x + bt.x, 0.f);
  o.y = fmaxf((v[1] - mean) * inv * g4.y + bt.y, 0.f);
  o.z = fmaxf((v[2] - mean) * inv * g4.z + bt.z, 0.f);
  o.w = fmaxf((v[3] - mean) * inv * g4.w + bt.w, 0.f);
  *(float4*)(out + base) = o;
}

// segment mean/max pooling accumulators (one wave per node)
__global__ void __launch_bounds__(256) k_pool(const float* __restrict__ h,
                                              const int* __restrict__ batch,
                                              float* __restrict__ gsum,
                                              float* __restrict__ gmax,
                                              float* __restrict__ gcnt, int n) {
  int node = blockIdx.x * 8 + (threadIdx.x >> 5);
  if (node >= n) return;
  int lane = threadIdx.x & 31;
  int g = batch[node];
  float4 v = *(const float4*)(h + (size_t)node * 128 + lane * 4);
  float* sp = gsum + (size_t)g * 128 + lane * 4;
  float* mp = gmax + (size_t)g * 128 + lane * 4;
  atomicAdd(sp + 0, v.x); atomicAdd(sp + 1, v.y);
  atomicAdd(sp + 2, v.z); atomicAdd(sp + 3, v.w);
  atomicMaxF(mp + 0, v.x); atomicMaxF(mp + 1, v.y);
  atomicMaxF(mp + 2, v.z); atomicMaxF(mp + 3, v.w);
  if (lane == 0) atomicAdd(&gcnt[g], 1.0f);
}

// whole graph head per block (64 blocks, negligible work)
__global__ void __launch_bounds__(256) k_graph_head(
    const float* __restrict__ gsum, const float* __restrict__ gmax,
    const float* __restrict__ gcnt,
    const float* __restrict__ gw1, const float* __restrict__ gb1,
    const float* __restrict__ glng, const float* __restrict__ glnb,
    const float* __restrict__ gw2, const float* __restrict__ gb2,
    const float* __restrict__ gw3, const float* __restrict__ gb3,
    float* __restrict__ out) {
  __shared__ float emb[256];
  __shared__ float z1[128];
  __shared__ float z2[64];
  __shared__ float red[2];
  int g = blockIdx.x;
  int t = threadIdx.x;
  float cnt = gcnt[g];
  if (t < 128) emb[t] = gsum[(size_t)g * 128 + t] / cnt;
  else         emb[t] = gmax[(size_t)g * 128 + (t - 128)];
  __syncthreads();
  if (t < 128) {
    float acc = gb1[t];
    for (int k = 0; k < 256; ++k) acc += emb[k] * gw1[(size_t)k * 128 + t];
    z1[t] = acc;
  }
  __syncthreads();
  if (t == 0) {
    float s = 0.f;
    for (int i = 0; i < 128; ++i) s += z1[i];
    red[0] = s * (1.0f / 128.0f);
  }
  __syncthreads();
  float mean = red[0];
  if (t == 0) {
    float s = 0.f;
    for (int i = 0; i < 128; ++i) { float d = z1[i] - mean; s += d * d; }
    red[1] = rsqrtf(s * (1.0f / 128.0f) + 1e-5f);
  }
  __syncthreads();
  if (t < 128) z1[t] = fmaxf((z1[t] - mean) * red[1] * glng[t] + glnb[t], 0.f);
  __syncthreads();
  if (t < 64) {
    float acc = gb2[t];
    for (int k = 0; k < 128; ++k) acc += z1[k] * gw2[(size_t)k * 64 + t];
    z2[t] = fmaxf(acc, 0.f);
  }
  __syncthreads();
  if (t < 2) {
    float acc = gb3[t];
    for (int k = 0; k < 64; ++k) acc += z2[k] * gw3[(size_t)k * 2 + t];
    out[g * 2 + t] = acc;
  }
}

__global__ void k_node_logits(const float* __restrict__ u2,
                              const float* __restrict__ nw3,
                              const float* __restrict__ nb3,
                              float* __restrict__ out, int n) {
  int i = blockIdx.x * blockDim.x + threadIdx.x;
  if (i >= n) return;
  const float* r = u2 + (size_t)i * 64;
  float a0 = nb3[0], a1 = nb3[1];
  for (int k = 0; k < 64; ++k) {
    float v = r[k];
    a0 += v * nw3[k * 2 + 0];
    a1 += v * nw3[k * 2 + 1];
  }
  out[i * 2 + 0] = a0;
  out[i * 2 + 1] = a1;
}

// ------------------------------------------------------------------ launcher

extern "C" void kernel_launch(void* const* d_in, const int* in_sizes, int n_in,
                              void* d_out, int out_size, void* d_ws, size_t ws_size,
                              hipStream_t stream) {
  const int Nn = in_sizes[2];        // nodes (batch length)
  const int Ee = in_sizes[1] / 2;    // edges

  const float* x     = (const float*)d_in[0];
  const int*   src   = (const int*)d_in[1];
  const int*   dst   = src + Ee;
  const int*   batch = (const int*)d_in[2];
  const float* w_in  = (const float*)d_in[3];
  const float* b_in  = (const float*)d_in[4];
  const float* convw = (const float*)d_in[5];
  const float* convb = (const float*)d_in[6];
  const float* lng   = (const float*)d_in[7];
  const float* lnb   = (const float*)d_in[8];
  const float* gw1 = (const float*)d_in[9],  *gb1 = (const float*)d_in[10];
  const float* glng= (const float*)d_in[11], *glnb= (const float*)d_in[12];
  const float* gw2 = (const float*)d_in[13], *gb2 = (const float*)d_in[14];
  const float* gw3 = (const float*)d_in[15], *gb3 = (const float*)d_in[16];
  const float* nw1 = (const float*)d_in[17], *nb1 = (const float*)d_in[18];
  const float* nlng= (const float*)d_in[19], *nlnb= (const float*)d_in[20];
  const float* nw2 = (const float*)d_in[21], *nb2 = (const float*)d_in[22];
  const float* nw3 = (const float*)d_in[23], *nb3 = (const float*)d_in[24];
  float* out = (float*)d_out;

  size_t off = 0;
  auto carve = [&](size_t bytes) -> char* {
    char* p = (char*)d_ws + off;
    off += (bytes + 255) & ~(size_t)255;
    return p;
  };
  float*    h    = (float*)carve((size_t)Nn * 128 * 4);
  float*    xw   = (float*)carve((size_t)Nn * 128 * 4);
  float*    agg  = (float*)carve((size_t)Nn * 128 * 4);
  bf16t*    hbf  = (bf16t*)carve((size_t)Nn * 128 * 2);
  float*    deg  = (float*)carve((size_t)Nn * 4);
  float*    dinv = (float*)carve((size_t)Nn * 4);
  float*    sfn  = (float*)carve((size_t)Nn * 4);
  float*    gsum = (float*)carve(64 * 128 * 4);
  float*    gmax = (float*)carve(64 * 128 * 4);
  float*    gcnt = (float*)carve(64 * 4);
  uint32_t* bpk  = (uint32_t*)carve(4 * 8 * 8 * 32 * 4);

  const int waves      = (Nn + 15) / 16;
  const int gemm_blk   = (waves + 7) / 8;
  const int node_blk   = (Nn + 7) / 8;
  const int edge_blk   = (Ee + 7) / 8;
  const int cvt_blk    = ((Nn * 128 / 4) + 255) / 256;
  const int pack8_blk  = (4 * 8 * 8 * 32 + 255) / 256;
  const int pack4_blk  = (4 * 4 * 8 * 32 + 255) / 256;

  // --- init accumulators
  hipMemsetAsync(deg, 0, (size_t)Nn * 4, stream);
  hipMemsetAsync(gsum, 0, 64 * 128 * 4, stream);
  hipMemsetAsync(gcnt, 0, 64 * 4, stream);
  k_fill<<<(64 * 128 + 255) / 256, 256, 0, stream>>>(gmax, -__builtin_huge_valf(), 64 * 128);

  // --- input projection: h = relu(x @ w_in + b_in)
  k_f2bf<<<cvt_blk, 256, 0, stream>>>(x, hbf, Nn * 128 / 4);
  k_pack_w<<<pack8_blk, 256, 0, stream>>>(w_in, bpk, 128);
  k_gemm<8><<<gemm_blk, 256, 0, stream>>>(hbf, bpk, b_in, h, Nn, 1);

  // --- degree / normalization
  k_deg<<<(Ee + 255) / 256, 256, 0, stream>>>(dst, deg, Ee);
  k_norm<<<(Nn + 255) / 256, 256, 0, stream>>>(deg, dinv, sfn, Nn);

  // --- 4 GCN layers
  for (int l = 0; l < 4; ++l) {
    k_f2bf<<<cvt_blk, 256, 0, stream>>>(h, hbf, Nn * 128 / 4);
    k_pack_w<<<pack8_blk, 256, 0, stream>>>(convw + (size_t)l * 128 * 128, bpk, 128);
    k_gemm<8><<<gemm_blk, 256, 0, stream>>>(hbf, bpk, nullptr, xw, Nn, 0);
    hipMemsetAsync(agg, 0, (size_t)Nn * 128 * 4, stream);
    k_scatter<<<edge_blk, 256, 0, stream>>>(src, dst, dinv, xw, agg, Ee);
    k_conv_post<<<node_blk, 256, 0, stream>>>(agg, xw, sfn, convb + l * 128,
                                              lng + l * 128, lnb + l * 128, h, Nn);
  }

  // --- pooling + graph head -> out[0:128]
  k_pool<<<node_blk, 256, 0, stream>>>(h, batch, gsum, gmax, gcnt, Nn);
  k_graph_head<<<64, 256, 0, stream>>>(gsum, gmax, gcnt, gw1, gb1, glng, glnb,
                                       gw2, gb2, gw3, gb3, out);

  // --- node head -> out[128:]
  k_f2bf<<<cvt_blk, 256, 0, stream>>>(h, hbf, Nn * 128 / 4);
  k_pack_w<<<pack8_blk, 256, 0, stream>>>(nw1, bpk, 128);
  k_gemm<8><<<gemm_blk, 256, 0, stream>>>(hbf, bpk, nullptr, xw, Nn, 0);   // u_pre
  k_ln_relu<<<node_blk, 256, 0, stream>>>(xw, nb1, nlng, nlnb, agg, Nn);   // u
  k_f2bf<<<cvt_blk, 256, 0, stream>>>(agg, hbf, Nn * 128 / 4);
  k_pack_w<<<pack4_blk, 256, 0, stream>>>(nw2, bpk, 64);
  k_gemm<4><<<gemm_blk, 256, 0, stream>>>(hbf, bpk, nb2, xw, Nn, 1);       // u2 [N,64]
  k_node_logits<<<(Nn + 255) / 256, 256, 0, stream>>>(xw, nw3, nb3, out + 128, Nn);
}
